// SmallMotionEncoder_57329223467424
// MI455X (gfx1250) — 16-bit and 8-bit matrix operand order repaired and checked by reading, compiled, not run on hardware
//
#include <hip/hip_runtime.h>
#include <hip/hip_bf16.h>
#include <math.h>

typedef __attribute__((ext_vector_type(16))) _Float16 v16h;
typedef __attribute__((ext_vector_type(8)))  float    v8f;
typedef __attribute__((ext_vector_type(8)))  int      v8i;

#define BATCH 8
#define H_    120
#define W_    160
#define HW    (H_*W_)
#define NPOS  (BATCH*HW)
#define CCORR 196
#define NT_C1 6
#define NK_C1 7
#define NT_P  5
#define NK_P  2
#define NTILES (NPOS/16)

#define BC1_BYTES ((size_t)NK_C1*NT_C1*1024)
#define BP_BYTES  ((size_t)NK_P*NT_P*1024)

static constexpr float ACT  = 6.0f     / 127.0f;
static constexpr float FLOW = 3.96875f / 127.0f;

static constexpr size_t OFF_SCALES = 0;
static constexpr size_t OFF_BC1    = 64;
static constexpr size_t OFF_BP     = OFF_BC1 + BC1_BYTES;
static constexpr size_t OFF_WF1    = OFF_BP  + BP_BYTES;
static constexpr size_t OFF_WF2    = OFF_WF1 + 512;
static constexpr size_t OFF_WD     = OFF_WF2 + 8192;
static constexpr size_t OFF_DWIN   = 66688;
static constexpr size_t OFF_PWIN   = OFF_DWIN + (size_t)NPOS*128;

__device__ __forceinline__ float qclamp(float x) {
    return fminf(fmaxf(rintf(x), -127.0f), 127.0f);
}

__device__ __forceinline__ unsigned packh2(float a, float b) {
    _Float16 ha = (_Float16)a, hb = (_Float16)b;
    unsigned short ua = __builtin_bit_cast(unsigned short, ha);
    unsigned short ub = __builtin_bit_cast(unsigned short, hb);
    return (unsigned)ua | ((unsigned)ub << 16);
}

__device__ __forceinline__ void async_g2l_b128(unsigned lds_off, const void* gaddr) {
    asm volatile("global_load_async_to_lds_b128 %0, %1, off"
                 :: "v"(lds_off), "v"(gaddr) : "memory");
}
__device__ __forceinline__ void wait_asynccnt0() {
    asm volatile("s_wait_asynccnt 0x0" ::: "memory");
}

__global__ void wprep(const float* __restrict__ Wc1, const float* __restrict__ Wp,
                      const float* __restrict__ Wf1, const float* __restrict__ Wf2,
                      const float* __restrict__ Wd, char* __restrict__ ws) {
    __shared__ float red[256];
    const float* Wt; int n;
    switch (blockIdx.x) {
        case 0: Wt = Wc1; n = 96*CCORR; break;
        case 1: Wt = Wp;  n = 80*128;   break;
        case 2: Wt = Wf1; n = 64*2;     break;
        case 3: Wt = Wf2; n = 32*64;    break;
        default:Wt = Wd;  n = 128*9;    break;
    }
    float m = 0.0f;
    for (int i = threadIdx.x; i < n; i += 256) m = fmaxf(m, fabsf(Wt[i]));
    red[threadIdx.x] = m;
    __syncthreads();
    for (int s = 128; s > 0; s >>= 1) {
        if (threadIdx.x < (unsigned)s) red[threadIdx.x] = fmaxf(red[threadIdx.x], red[threadIdx.x+s]);
        __syncthreads();
    }
    float scale = red[0] / 127.0f + 1e-12f;
    if (threadIdx.x == 0) ((float*)(ws + OFF_SCALES))[blockIdx.x] = scale;

    if (blockIdx.x == 0) {
        unsigned* dst = (unsigned*)(ws + OFF_BC1);
        for (int idx = threadIdx.x; idx < NK_C1*NT_C1*32*8; idx += 256) {
            int v = idx & 7, lane = (idx >> 3) & 31, pair = idx >> 8;
            int ks = pair / NT_C1, nt = pair % NT_C1;
            int khalf = lane >> 4;
            int nn = nt*16 + (lane & 15);
            int k0 = ks*32 + khalf*16 + 2*v;
            float q0 = (k0   < CCORR) ? qclamp(Wc1[nn*CCORR + k0]     / scale) : 0.0f;
            float q1 = (k0+1 < CCORR) ? qclamp(Wc1[nn*CCORR + k0 + 1] / scale) : 0.0f;
            dst[(size_t)pair*256 + lane*8 + v] = packh2(q0, q1);
        }
    } else if (blockIdx.x == 1) {
        unsigned* dst = (unsigned*)(ws + OFF_BP);
        for (int idx = threadIdx.x; idx < NK_P*NT_P*32*8; idx += 256) {
            int v = idx & 7, lane = (idx >> 3) & 31, pair = idx >> 8;
            int ks = pair / NT_P, nt = pair % NT_P;
            int khalf = lane >> 4;
            int nn = nt*16 + (lane & 15);
            int kb = ks*64 + ((v & 4) ? 32 : 0) + khalf*16 + (v & 3)*4;
            unsigned w = 0;
            for (int j = 0; j < 4; j++) {
                int q = (int)qclamp(Wp[nn*128 + kb + j] / scale);
                w |= ((unsigned)(q & 0xff)) << (8*j);
            }
            dst[(size_t)pair*256 + lane*8 + v] = w;
        }
    } else if (blockIdx.x == 2) {
        float* dst = (float*)(ws + OFF_WF1);
        for (int i = threadIdx.x; i < 128; i += 256)  dst[i] = scale * qclamp(Wf1[i] / scale);
    } else if (blockIdx.x == 3) {
        float* dst = (float*)(ws + OFF_WF2);
        for (int i = threadIdx.x; i < 2048; i += 256) dst[i] = scale * qclamp(Wf2[i] / scale);
    } else {
        float* dst = (float*)(ws + OFF_WD);
        for (int i = threadIdx.x; i < 1152; i += 256) dst[i] = scale * qclamp(Wd[i] / scale);
    }
}

__global__ __launch_bounds__(256) void convc1_kernel(const float* __restrict__ corr,
                                                     const float* __restrict__ bc1,
                                                     char* __restrict__ ws) {
    __shared__ __align__(32) char smB[BC1_BYTES];

    {
        const char* src = ws + OFF_BC1;
        for (int i = threadIdx.x; i < (int)(BC1_BYTES / 16); i += 256) {
            unsigned dst = (unsigned)(size_t)(&smB[i * 16]);
            async_g2l_b128(dst, src + (size_t)i * 16);
        }
        wait_asynccnt0();
        __syncthreads();
    }

    int wave = (blockIdx.x * blockDim.x + threadIdx.x) >> 5;
    int lane = threadIdx.x & 31;
    int half = lane >> 4, mrow = lane & 15;
    int tile = wave;

    int posA = tile*16 + mrow;
    int bA = posA / HW, hwA = posA % HW;
    const float* corrBase = corr + (size_t)bA * CCORR * HW + hwA;

    v8f acc[NT_C1];
    #pragma unroll
    for (int nt = 0; nt < NT_C1; nt++) acc[nt] = (v8f)0.0f;

    for (int ks = 0; ks < NK_C1; ks++) {
        v16h a;
        int kbase = ks*32 + half*16;
        #pragma unroll
        for (int j = 0; j < 8; j++) {
            int k  = kbase + j;
            int k2 = kbase + 8 + j;
            float x  = (k  < CCORR) ? __builtin_nontemporal_load(corrBase + (size_t)k  * HW) : 0.0f;
            float x2 = (k2 < CCORR) ? __builtin_nontemporal_load(corrBase + (size_t)k2 * HW) : 0.0f;
            a[j]     = (_Float16)x;
            a[8 + j] = (_Float16)x2;
        }
        #pragma unroll
        for (int nt = 0; nt < NT_C1; nt++) {
            const v16h b = *(const v16h*)(smB + ((size_t)((ks*NT_C1 + nt)*32 + lane)) * 32);
            acc[nt] = __builtin_amdgcn_wmma_f32_16x16x32_f16(false, a, false, b,
                                                             (short)0, acc[nt], false, false);
        }
    }

    float s1 = ((const float*)(ws + OFF_SCALES))[0];
    signed char* dwin = (signed char*)(ws + OFF_DWIN);
    #pragma unroll
    for (int nt = 0; nt < NT_C1; nt++) {
        #pragma unroll
        for (int r = 0; r < 8; r++) {
            int M   = r + half*8;
            int pos = tile*16 + M;
            int ch  = nt*16 + mrow;
            float y  = s1 * acc[nt][r] + bc1[ch];
            float q  = fmaxf(qclamp(y / ACT), 0.0f);
            float c  = q * ACT;
            float qf = qclamp(c / FLOW);
            float cf = qf * FLOW;
            float cd = qclamp(cf / ACT);
            dwin[(size_t)pos*128 + ch] = (signed char)cd;
        }
    }
}

__global__ void flowk(const float* __restrict__ flow, const float* __restrict__ bf1,
                      const float* __restrict__ bf2, char* __restrict__ ws,
                      float* __restrict__ out) {
    int p = blockIdx.x * blockDim.x + threadIdx.x;
    if (p >= NPOS) return;
    int b = p / HW, hw = p % HW;
    float f0 = flow[(size_t)b*2*HW + hw];
    float f1 = flow[(size_t)b*2*HW + HW + hw];
    const float* w1 = (const float*)(ws + OFF_WF1);
    const float* w2 = (const float*)(ws + OFF_WF2);

    float a[64];
    #pragma unroll 8
    for (int c = 0; c < 64; c++) {
        float h = fmaxf(w1[c*2]*f0 + w1[c*2+1]*f1 + bf1[c], 0.0f);
        a[c] = qclamp(h / ACT) * ACT;
    }
    signed char* dwin = (signed char*)(ws + OFF_DWIN);
    for (int o = 0; o < 32; o++) {
        float h = bf2[o];
        #pragma unroll 8
        for (int i = 0; i < 64; i++) h += a[i] * w2[o*64 + i];
        float q  = fmaxf(qclamp(h / ACT), 0.0f);
        float fl = q * ACT;
        float qf = qclamp(fl / FLOW);
        float ff = qf * FLOW;
        float cd = qclamp(ff / ACT);
        dwin[(size_t)p*128 + 96 + o] = (signed char)cd;
    }
    #pragma unroll
    for (int c = 0; c < 2; c++) {
        float fv = flow[(size_t)b*2*HW + c*HW + hw];
        __builtin_nontemporal_store(qclamp(fv / FLOW) * FLOW,
                                    out + (size_t)b*82*HW + (size_t)(80 + c)*HW + hw);
    }
}

__global__ void dwk(const float* __restrict__ bd, char* __restrict__ ws) {
    size_t t = (size_t)blockIdx.x * 256 + threadIdx.x;
    int ch = (int)(t & 127);
    size_t pos = t >> 7;
    int b = (int)(pos / HW), hw = (int)(pos % HW);
    int h = hw / W_, w = hw % W_;
    const signed char* dwin = (const signed char*)(ws + OFF_DWIN);
    const float* wd = (const float*)(ws + OFF_WD);
    float acc = 0.0f;
    #pragma unroll
    for (int dy = -1; dy <= 1; dy++) {
        #pragma unroll
        for (int dx = -1; dx <= 1; dx++) {
            int hh = h + dy, ww = w + dx;
            if (hh >= 0 && hh < H_ && ww >= 0 && ww < W_) {
                int q = dwin[((size_t)b*HW + (size_t)hh*W_ + ww)*128 + ch];
                acc += (float)q * wd[ch*9 + (dy+1)*3 + (dx+1)];
            }
        }
    }
    float d = ACT * acc + bd[ch];
    signed char* pwin = (signed char*)(ws + OFF_PWIN);
    pwin[pos*128 + ch] = (signed char)qclamp(d / ACT);
}

__global__ __launch_bounds__(256) void convp_kernel(const float* __restrict__ bp,
                                                    char* __restrict__ ws,
                                                    float* __restrict__ out) {
    __shared__ __align__(32) char smB[BP_BYTES];

    {
        const char* src = ws + OFF_BP;
        for (int i = threadIdx.x; i < (int)(BP_BYTES / 16); i += 256) {
            unsigned dst = (unsigned)(size_t)(&smB[i * 16]);
            async_g2l_b128(dst, src + (size_t)i * 16);
        }
        wait_asynccnt0();
        __syncthreads();
    }

    int wave = (blockIdx.x * blockDim.x + threadIdx.x) >> 5;
    int lane = threadIdx.x & 31;
    int half = lane >> 4, mrow = lane & 15;
    int tile = wave;

    const char* pwin = ws + OFF_PWIN;
    const char* abase = pwin + (size_t)(tile*16 + mrow)*128 + half*16;

    v8i acc[NT_P];
    #pragma unroll
    for (int nt = 0; nt < NT_P; nt++) acc[nt] = (v8i)0;

    #pragma unroll
    for (int ks = 0; ks < NK_P; ks++) {
        v8i a;
        #pragma unroll
        for (int blk = 0; blk < 4; blk++) {
            a[2*blk]     = *(const int*)(abase + ks*64 + blk*8 + (blk>>1)*16);
            a[2*blk + 1] = *(const int*)(abase + ks*64 + blk*8 + (blk>>1)*16 + 4);
        }
        #pragma unroll
        for (int nt = 0; nt < NT_P; nt++) {
            const v8i bm = *(const v8i*)(smB + ((size_t)((ks*NT_P + nt)*32 + lane)) * 32);
            acc[nt] = __builtin_amdgcn_wmma_i32_16x16x64_iu8(true, a, true, bm,
                                                             acc[nt], false, false);
        }
    }

    float sp = ((const float*)(ws + OFF_SCALES))[1];
    float ds = ACT * sp;
    #pragma unroll
    for (int nt = 0; nt < NT_P; nt++) {
        #pragma unroll
        for (int r = 0; r < 8; r++) {
            int M   = r + half*8;
            int pos = tile*16 + M;
            int ch  = nt*16 + mrow;
            int b = pos / HW, hw = pos % HW;
            float y = fmaxf(ds * (float)acc[nt][r] + bp[ch], 0.0f);
            __builtin_nontemporal_store(qclamp(y / FLOW) * FLOW,
                                        out + (size_t)b*82*HW + (size_t)ch*HW + hw);
        }
    }
}

extern "C" void kernel_launch(void* const* d_in, const int* in_sizes, int n_in,
                              void* d_out, int out_size, void* d_ws, size_t ws_size,
                              hipStream_t stream) {
    const float* flow = (const float*)d_in[0];
    const float* corr = (const float*)d_in[1];
    const float* Wc1  = (const float*)d_in[2];
    const float* bc1  = (const float*)d_in[3];
    const float* Wf1  = (const float*)d_in[4];
    const float* bf1  = (const float*)d_in[5];
    const float* Wf2  = (const float*)d_in[6];
    const float* bf2  = (const float*)d_in[7];
    const float* Wd   = (const float*)d_in[8];
    const float* bd   = (const float*)d_in[9];
    const float* Wp   = (const float*)d_in[10];
    const float* bp   = (const float*)d_in[11];
    float* out = (float*)d_out;
    char* ws = (char*)d_ws;

    wprep<<<5, 256, 0, stream>>>(Wc1, Wp, Wf1, Wf2, Wd, ws);
    convc1_kernel<<<NTILES/8, 256, 0, stream>>>(corr, bc1, ws);
    flowk<<<NPOS/256, 256, 0, stream>>>(flow, bf1, bf2, ws, out);
    dwk<<<(NPOS*128)/256, 256, 0, stream>>>(bd, ws);
    convp_kernel<<<NTILES/8, 256, 0, stream>>>(bp, ws, out);
}
